// MultiHeadAttention_66571993088136
// MI455X (gfx1250) — matrix operand order repaired and checked by reading, not compiled, not run on hardware
//
#include <hip/hip_runtime.h>
#include <hip/hip_bf16.h>

// ---------------------------------------------------------------------------
// MI455X (gfx1250) causal multi-head attention, bf16 WMMA pipeline.
//   B=1, S=2048, D=4096, H=32, HD=128.
// Compute-bound (~344 GFLOP vs ~14us of HBM traffic at 23.3 TB/s), so all
// GEMMs run on v_wmma_f32_16x16x32_bf16 with f32 accumulation.
// This revision software-pipelines the GEMM inner loops (grouped fragment
// loads into distinct registers so s_wait_loadcnt can retire incrementally)
// and uses temporal (cache-filling) prefetch for the streamed weights.
// ---------------------------------------------------------------------------

#define S_LEN 2048
#define D_DIM 4096
#define NH    32
#define HD    128

typedef __bf16 bf16_t;
typedef __attribute__((ext_vector_type(16))) __bf16 v16bf;
typedef __attribute__((ext_vector_type(8)))  __bf16 v8bf;
typedef __attribute__((ext_vector_type(4)))  __bf16 v4bf;
typedef __attribute__((ext_vector_type(8)))  float  v8f;
typedef __attribute__((ext_vector_type(4)))  float  v4f;

static __device__ __forceinline__ bf16_t to_bf16(float f) {
  unsigned u = __builtin_bit_cast(unsigned, f);
  unsigned r = u + 0x7FFFu + ((u >> 16) & 1u);   // round-to-nearest-even
  unsigned short s = (unsigned short)(r >> 16);
  return __builtin_bit_cast(bf16_t, s);
}

static __device__ __forceinline__ v8f vzero8() {
  v8f z;
#pragma unroll
  for (int i = 0; i < 8; ++i) z[i] = 0.f;
  return z;
}

// A-operand (16x32 bf16, MxK). Layout: lane m=L&15, lo=16*(L>>4);
// elements = K[lo..lo+15] contiguous (matches load_b_frag).
static __device__ __forceinline__ v16bf load_a_frag(const bf16_t* __restrict__ p, int ld) {
  const int lane = threadIdx.x & 31;
  const int m    = lane & 15;
  const int lo   = (lane >> 4) << 4;             // 0 or 16
  const bf16_t* row = p + (size_t)m * ld + lo;
  return *(const v16bf*)(row);                   // K = lo .. lo+15
}

// B-operand (32x16 bf16, KxN), column n = row n of a row-major [N x K] source.
// Same k order as load_a_frag: lane n=L&15 holds K = 16*(L>>4) .. +15 contiguous.
static __device__ __forceinline__ v16bf load_b_frag(const bf16_t* __restrict__ p, int ld) {
  const int lane = threadIdx.x & 31;
  const int n    = lane & 15;
  const int kh   = (lane >> 4) << 4;             // 0 or 16
  return *(const v16bf*)(p + (size_t)n * ld + kh);
}

static __device__ __forceinline__ v8f wmma_bf16(v16bf a, v16bf b, v8f c) {
  // (neg_a, A, neg_b, B, c_mod, C, reuse_a, reuse_b)
  return __builtin_amdgcn_wmma_f32_16x16x32_bf16(false, a, false, b, (short)0, c,
                                                 false, false);
}

// Row reductions within 16-lane halves (C-layout rows live per-half).
static __device__ __forceinline__ float half_reduce_max(float x) {
  x = fmaxf(x, __shfl_xor(x, 1, 16));
  x = fmaxf(x, __shfl_xor(x, 2, 16));
  x = fmaxf(x, __shfl_xor(x, 4, 16));
  x = fmaxf(x, __shfl_xor(x, 8, 16));
  return x;
}
static __device__ __forceinline__ float half_reduce_sum(float x) {
  x += __shfl_xor(x, 1, 16);
  x += __shfl_xor(x, 2, 16);
  x += __shfl_xor(x, 4, 16);
  x += __shfl_xor(x, 8, 16);
  return x;
}

// ---------------------------------------------------------------------------
// Kernel 1: f32 -> bf16 conversion (vectorized x4).
// ---------------------------------------------------------------------------
__global__ void __launch_bounds__(256) cvt_bf16_kernel(const float* __restrict__ in,
                                                       bf16_t* __restrict__ out, int n4) {
  int i = blockIdx.x * blockDim.x + threadIdx.x;
  if (i >= n4) return;
  v4f f = *(const v4f*)(in + 4 * (size_t)i);
  v4bf o;
#pragma unroll
  for (int j = 0; j < 4; ++j) o[j] = to_bf16(f[j]);
  *(v4bf*)(out + 4 * (size_t)i) = o;
}

// ---------------------------------------------------------------------------
// Kernel 2: QKV projection (X @ W^T), RoPE fused for Q/K, V stored transposed.
// One wave (32 thr) computes a 16(s) x 128(one head) tile of one of Q/K/V.
// ---------------------------------------------------------------------------
__global__ void __launch_bounds__(32) qkv_rope_kernel(
    const bf16_t* __restrict__ Xb,
    const bf16_t* __restrict__ Wqb, const bf16_t* __restrict__ Wkb,
    const bf16_t* __restrict__ Wvb,
    const float* __restrict__ cosT, const float* __restrict__ sinT,
    bf16_t* __restrict__ Q, bf16_t* __restrict__ Kout, bf16_t* __restrict__ Vt) {
  const int wid = blockIdx.x;
  const int mat = wid / (NH * (S_LEN / 16));
  const int rem = wid % (NH * (S_LEN / 16));
  const int h   = rem / (S_LEN / 16);
  const int sb  = rem % (S_LEN / 16);
  const int s0  = sb * 16;
  const int n0  = h * HD;
  const bf16_t* W = (mat == 0) ? Wqb : (mat == 1) ? Wkb : Wvb;

  const int lane = threadIdx.x & 31;
  const int c    = lane & 15;
  const int hh   = lane >> 4;

  v8f acc[8];
#pragma unroll
  for (int jc = 0; jc < 8; ++jc) acc[jc] = vzero8();

#pragma unroll 2
  for (int kt = 0; kt < D_DIM / 32; ++kt) {
    const int kb = kt * 32;
    if (kt + 2 < D_DIM / 32) {                     // temporal prefetch, 2 iters ahead
      __builtin_prefetch(Xb + (size_t)s0 * D_DIM + kb + 64, 0, 2);
      __builtin_prefetch(W + (size_t)n0 * D_DIM + kb + 64, 0, 2);
    }
    // Group all fragment loads into distinct registers first so the
    // scheduler can clause them and retire WMMAs with partial waits.
    v16bf a = load_a_frag(Xb + (size_t)s0 * D_DIM + kb, D_DIM);
    v16bf b[8];
#pragma unroll
    for (int jc = 0; jc < 8; ++jc)
      b[jc] = load_b_frag(W + (size_t)(n0 + 16 * jc) * D_DIM + kb, D_DIM);
#pragma unroll
    for (int jc = 0; jc < 8; ++jc)
      acc[jc] = wmma_bf16(a, b[jc], acc[jc]);
  }

  if (mat < 2) {
    // RoPE: out[j] = x[j]*cos - x[j+64]*sin ; out[j+64] = x[j+64]*cos + x[j]*sin
    bf16_t* Out = (mat == 0) ? Q : Kout;
#pragma unroll
    for (int jc = 0; jc < 4; ++jc) {
#pragma unroll
      for (int r = 0; r < 8; ++r) {
        const int srow = s0 + r + 8 * hh;
        const int j1   = 16 * jc + c;
        const float cv1 = cosT[srow * HD + j1];
        const float sv1 = sinT[srow * HD + j1];
        const float cv2 = cosT[srow * HD + j1 + 64];
        const float sv2 = sinT[srow * HD + j1 + 64];
        const float x1 = acc[jc][r];
        const float x2 = acc[jc + 4][r];
        Out[(size_t)srow * D_DIM + n0 + j1]      = to_bf16(x1 * cv1 - x2 * sv1);
        Out[(size_t)srow * D_DIM + n0 + j1 + 64] = to_bf16(x2 * cv2 + x1 * sv2);
      }
    }
  } else {
    // V stored transposed per head: Vt[(h*HD + j) * S + s]
#pragma unroll
    for (int jc = 0; jc < 8; ++jc) {
#pragma unroll
      for (int r = 0; r < 8; ++r) {
        const int srow = s0 + r + 8 * hh;
        const int j    = 16 * jc + c;
        Vt[(size_t)(n0 + j) * S_LEN + srow] = to_bf16(acc[jc][r]);
      }
    }
  }
}

// ---------------------------------------------------------------------------
// Kernel 3: causal flash attention. One wave per (head, 16-query block).
// ---------------------------------------------------------------------------
__global__ void __launch_bounds__(32) attn_kernel(const bf16_t* __restrict__ Qm,
                                                  const bf16_t* __restrict__ Km,
                                                  const bf16_t* __restrict__ Vt,
                                                  bf16_t* __restrict__ Om) {
  __shared__ __align__(32) bf16_t Plds[16 * 32];   // P tile bounce buffer (1 KB)
  const float SCORE_SCALE = 0.08838834764831845f;  // 1/sqrt(128)

  const int wid = blockIdx.x;
  const int h   = wid / (S_LEN / 16);
  const int qb  = wid % (S_LEN / 16);
  const int q0  = qb * 16;
  const int lane = threadIdx.x & 31;
  const int c    = lane & 15;
  const int hh   = lane >> 4;

  v16bf qa[4];
#pragma unroll
  for (int kk = 0; kk < 4; ++kk)
    qa[kk] = load_a_frag(Qm + (size_t)q0 * D_DIM + h * HD + 32 * kk, D_DIM);

  v8f acc[8];
#pragma unroll
  for (int jc = 0; jc < 8; ++jc) acc[jc] = vzero8();
  float m_run[8], l_run[8];
#pragma unroll
  for (int r = 0; r < 8; ++r) { m_run[r] = -__builtin_inff(); l_run[r] = 0.f; }

  const int nkb = (q0 + 16 + 31) >> 5;             // 32-key blocks, causal bound
  for (int kb = 0; kb < nkb; ++kb) {
    const int k0 = kb * 32;

    // ---- scores: load all 8 K fragments first, then 8 WMMAs ----
    v16bf kf[8];
#pragma unroll
    for (int kk = 0; kk < 4; ++kk) {
      kf[2 * kk]     = load_b_frag(Km + (size_t)k0 * D_DIM + h * HD + 32 * kk, D_DIM);
      kf[2 * kk + 1] = load_b_frag(Km + (size_t)(k0 + 16) * D_DIM + h * HD + 32 * kk, D_DIM);
    }
    v8f s0 = vzero8(), s1 = vzero8();
#pragma unroll
    for (int kk = 0; kk < 4; ++kk) {
      s0 = wmma_bf16(qa[kk], kf[2 * kk], s0);
      s1 = wmma_bf16(qa[kk], kf[2 * kk + 1], s1);
    }

    // scale + causal mask + online softmax (C layout: row = r + 8*hh, col = c)
    float esc[8];
#pragma unroll
    for (int r = 0; r < 8; ++r) {
      const int qrow = q0 + r + 8 * hh;
      float a0 = (k0 + c      <= qrow) ? s0[r] * SCORE_SCALE : -__builtin_inff();
      float a1 = (k0 + 16 + c <= qrow) ? s1[r] * SCORE_SCALE : -__builtin_inff();
      float bm = half_reduce_max(fmaxf(a0, a1));
      float mn = fmaxf(m_run[r], bm);
      float sc = __expf(m_run[r] - mn);
      float p0 = __expf(a0 - mn);
      float p1 = __expf(a1 - mn);
      s0[r] = p0; s1[r] = p1;
      l_run[r] = l_run[r] * sc + half_reduce_sum(p0 + p1);
      m_run[r] = mn;
      esc[r]   = sc;
    }
#pragma unroll
    for (int jc = 0; jc < 8; ++jc)
#pragma unroll
      for (int r = 0; r < 8; ++r) acc[jc][r] *= esc[r];

    // C-layout -> A-operand relayout of P through LDS (single-wave block:
    // only the DS counter needs draining, no barrier resource required).
#pragma unroll
    for (int r = 0; r < 8; ++r) {
      Plds[(r + 8 * hh) * 32 + c]      = to_bf16(s0[r]);
      Plds[(r + 8 * hh) * 32 + 16 + c] = to_bf16(s1[r]);
    }
    asm volatile("s_wait_dscnt 0" ::: "memory");

    // ---- PV: load all 8 V fragments first, then 8 accumulating WMMAs ----
    v16bf vf[8];
#pragma unroll
    for (int jc = 0; jc < 8; ++jc)
      vf[jc] = load_b_frag(Vt + (size_t)(h * HD + 16 * jc) * S_LEN + k0, S_LEN);
    v16bf pf = load_a_frag(Plds, 32);
#pragma unroll
    for (int jc = 0; jc < 8; ++jc)
      acc[jc] = wmma_bf16(pf, vf[jc], acc[jc]);
    asm volatile("s_wait_dscnt 0" ::: "memory");   // WAR vs next iter's P store
  }

  float inv[8];
#pragma unroll
  for (int r = 0; r < 8; ++r) inv[r] = 1.f / l_run[r];
#pragma unroll
  for (int jc = 0; jc < 8; ++jc)
#pragma unroll
    for (int r = 0; r < 8; ++r)
      Om[(size_t)(q0 + r + 8 * hh) * D_DIM + h * HD + 16 * jc + c] =
          to_bf16(acc[jc][r] * inv[r]);
}

// ---------------------------------------------------------------------------
// Kernel 4: output projection Y = O @ Wo^T + bo (f32 out).
// ---------------------------------------------------------------------------
__global__ void __launch_bounds__(32) oproj_kernel(const bf16_t* __restrict__ Ob,
                                                   const bf16_t* __restrict__ Wob,
                                                   const float* __restrict__ bo,
                                                   float* __restrict__ Y) {
  const int wid = blockIdx.x;
  const int nb  = wid % (D_DIM / HD);
  const int sb  = wid / (D_DIM / HD);
  const int s0  = sb * 16;
  const int n0  = nb * HD;
  const int lane = threadIdx.x & 31;
  const int c    = lane & 15;
  const int hh   = lane >> 4;

  v8f acc[8];
#pragma unroll
  for (int jc = 0; jc < 8; ++jc) acc[jc] = vzero8();

#pragma unroll 2
  for (int kt = 0; kt < D_DIM / 32; ++kt) {
    const int kb = kt * 32;
    if (kt + 2 < D_DIM / 32) {
      __builtin_prefetch(Ob + (size_t)s0 * D_DIM + kb + 64, 0, 2);
      __builtin_prefetch(Wob + (size_t)n0 * D_DIM + kb + 64, 0, 2);
    }
    v16bf a = load_a_frag(Ob + (size_t)s0 * D_DIM + kb, D_DIM);
    v16bf b[8];
#pragma unroll
    for (int jc = 0; jc < 8; ++jc)
      b[jc] = load_b_frag(Wob + (size_t)(n0 + 16 * jc) * D_DIM + kb, D_DIM);
#pragma unroll
    for (int jc = 0; jc < 8; ++jc)
      acc[jc] = wmma_bf16(a, b[jc], acc[jc]);
  }
#pragma unroll
  for (int jc = 0; jc < 8; ++jc)
#pragma unroll
    for (int r = 0; r < 8; ++r) {
      const int col = n0 + 16 * jc + c;
      Y[(size_t)(s0 + r + 8 * hh) * D_DIM + col] = acc[jc][r] + bo[col];
    }
}

// ---------------------------------------------------------------------------
// Host-side launcher. Scratch layout (bytes) in d_ws:
//   Xb 16M | Wq 32M | Wk 32M | Wv 32M | Wo 32M | Q 16M | K 16M | Vt 16M | O 16M
// ---------------------------------------------------------------------------
extern "C" void kernel_launch(void* const* d_in, const int* in_sizes, int n_in,
                              void* d_out, int out_size, void* d_ws, size_t ws_size,
                              hipStream_t stream) {
  (void)in_sizes; (void)n_in; (void)out_size; (void)ws_size;

  const float* X   = (const float*)d_in[0];
  const float* Wq  = (const float*)d_in[1];
  const float* Wk  = (const float*)d_in[2];
  const float* Wv  = (const float*)d_in[3];
  const float* Wo  = (const float*)d_in[4];
  const float* bo  = (const float*)d_in[5];
  const float* cosT = (const float*)d_in[6];
  const float* sinT = (const float*)d_in[7];
  float* Y = (float*)d_out;

  constexpr size_t SZ_X = (size_t)S_LEN * D_DIM * sizeof(bf16_t);   // 16 MiB
  constexpr size_t SZ_W = (size_t)D_DIM * D_DIM * sizeof(bf16_t);   // 32 MiB
  char* ws = (char*)d_ws;
  bf16_t* Xb  = (bf16_t*)(ws);
  bf16_t* Wqb = (bf16_t*)(ws + SZ_X);
  bf16_t* Wkb = (bf16_t*)(ws + SZ_X + SZ_W);
  bf16_t* Wvb = (bf16_t*)(ws + SZ_X + 2 * SZ_W);
  bf16_t* Wob = (bf16_t*)(ws + SZ_X + 3 * SZ_W);
  bf16_t* Qb  = (bf16_t*)(ws + SZ_X + 4 * SZ_W);
  bf16_t* Kb  = (bf16_t*)(ws + 2 * SZ_X + 4 * SZ_W);
  bf16_t* Vtb = (bf16_t*)(ws + 3 * SZ_X + 4 * SZ_W);
  bf16_t* Obf = (bf16_t*)(ws + 4 * SZ_X + 4 * SZ_W);

  // 1) f32 -> bf16 conversions
  {
    const int nx4 = (S_LEN * D_DIM) / 4;        // 2,097,152
    const int nw4 = (D_DIM * D_DIM) / 4;        // 4,194,304
    cvt_bf16_kernel<<<(nx4 + 255) / 256, 256, 0, stream>>>(X,  Xb,  nx4);
    cvt_bf16_kernel<<<(nw4 + 255) / 256, 256, 0, stream>>>(Wq, Wqb, nw4);
    cvt_bf16_kernel<<<(nw4 + 255) / 256, 256, 0, stream>>>(Wk, Wkb, nw4);
    cvt_bf16_kernel<<<(nw4 + 255) / 256, 256, 0, stream>>>(Wv, Wvb, nw4);
    cvt_bf16_kernel<<<(nw4 + 255) / 256, 256, 0, stream>>>(Wo, Wob, nw4);
  }

  // 2) QKV projection + RoPE (+ V transpose): 3 mats * 32 heads * 128 s-blocks
  qkv_rope_kernel<<<3 * NH * (S_LEN / 16), 32, 0, stream>>>(
      Xb, Wqb, Wkb, Wvb, cosT, sinT, Qb, Kb, Vtb);

  // 3) Flash attention: 32 heads * 128 query blocks
  attn_kernel<<<NH * (S_LEN / 16), 32, 0, stream>>>(Qb, Kb, Vtb, Obf);

  // 4) Output projection + bias
  oproj_kernel<<<(S_LEN / 16) * (D_DIM / HD), 32, 0, stream>>>(Obf, Wob, bo, Y);
}
